// MultiHeadAttentionModule_79199196938658
// MI455X (gfx1250) — matrix operand order repaired and checked by reading, not compiled, not run on hardware
//
#include <hip/hip_runtime.h>
#include <hip/hip_bf16.h>

// ---------------------------------------------------------------------------
// MI455X (gfx1250) multi-head attention: 1x1-conv QKV proj + flash attention.
// All GEMMs on v_wmma_f32_16x16x32_bf16 (fp32 accumulate), wave32 layouts
// per CDNA5 ISA 7.12.2. Scores never hit memory (online softmax).
// ---------------------------------------------------------------------------

typedef __attribute__((ext_vector_type(16))) __bf16 v16bf;
typedef __attribute__((ext_vector_type(8)))  __bf16 v8bf;
typedef __attribute__((ext_vector_type(8)))  float  v8f;

#define NB     4
#define CIN    256
#define COUT   256
#define NPIX   3136          // 56*56
#define HEADS  8
#define HD     32
#define SCALE  0.17677669529663687f   // 1/sqrt(32)

// workspace layout (bf16 element offsets); total = 26,083,328 bytes
#define XT_OFF    0                        // Xt  [B][N][CIN]   (x transposed, bf16)
#define XT_ELEMS  (NB*NPIX*CIN)            // 3,211,264
#define W_OFF     (XT_OFF + XT_ELEMS)      // Wq|Wk|Wv [COUT][CIN] bf16
#define W_ELEMS   (3*COUT*CIN)             // 196,608
#define Q_OFF     (W_OFF + W_ELEMS)        // Q [B][h][N][HD]
#define QKV_ELEMS (NB*HEADS*NPIX*HD)       // 3,211,264
#define K_OFF     (Q_OFF + QKV_ELEMS)      // K [B][h][N][HD]
#define V_OFF     (K_OFF + QKV_ELEMS)      // Vt [B][h][HD][N]  (transposed for PV B-frags)

__device__ __forceinline__ v8f wmma_bf16(v16bf a, v16bf b, v8f c) {
    return __builtin_amdgcn_wmma_f32_16x16x32_bf16(
        /*neg_a=*/false, a, /*neg_b=*/false, b,
        /*c_mod=*/(short)0, c, /*reuse_a=*/false, /*reuse_b=*/false);
}

// ---------------------------------------------------------------------------
// Kernel 1: fp32 -> bf16 conversion; x gets transposed to [B][N][CIN] so every
// later WMMA fragment is two contiguous b128 loads per lane.
// ---------------------------------------------------------------------------
__global__ void cvt_kernel(const float* __restrict__ x,
                           const float* __restrict__ wq,
                           const float* __restrict__ wk,
                           const float* __restrict__ wv,
                           __bf16* __restrict__ ws) {
    int idx = blockIdx.x * 256 + threadIdx.x;
    if (idx < XT_ELEMS) {
        int b = idx / (NPIX * CIN);
        int r = idx % (NPIX * CIN);
        int n = r / CIN;
        int c = r % CIN;
        ws[XT_OFF + idx] = (__bf16)x[(b * CIN + c) * NPIX + n];
    } else {
        int widx = idx - XT_ELEMS;
        if (widx < W_ELEMS) {
            const float* src = (widx < COUT*CIN) ? wq
                             : (widx < 2*COUT*CIN) ? wk : wv;
            ws[W_OFF + widx] = (__bf16)src[widx % (COUT*CIN)];
        }
    }
}

// ---------------------------------------------------------------------------
// Kernel 2: QKV projection.  D[n,m] = sum_k Xt[n,k]*W[m,k] + bias[m].
// One wave = 16 pixels x 64 channels, K=256 in 8 WMMA steps (32 WMMAs/wave).
// A-frag (16-bit 16x32): lanes 0-15 hold K 0-15, lanes 16-31 K 16-31
// of their row.  B-frag: lane column m, 16 contiguous K per half-wave.
// Q,K stored [B][h][N][HD]; V stored transposed [B][h][HD][N].
// ---------------------------------------------------------------------------
__global__ __launch_bounds__(256) void proj_kernel(__bf16* ws,
                                                   const float* __restrict__ bq,
                                                   const float* __restrict__ bk,
                                                   const float* __restrict__ bv) {
    const int lane  = threadIdx.x & 31;
    const int wave  = threadIdx.x >> 5;
    const int t     = blockIdx.x * 8 + wave;     // 9408 tiles total (exact)
    if (t >= 3 * NB * (NPIX/16) * (COUT/64)) return;

    const int p    = t / 3136;                   // 0=Q 1=K 2=V
    const int rem  = t % 3136;
    const int b    = rem / 784;
    const int rem2 = rem % 784;
    const int px0  = (rem2 / 4) * 16;
    const int c0   = (rem2 % 4) * 64;

    const __bf16* Xt  = ws + XT_OFF + (size_t)b * NPIX * CIN;
    const __bf16* Wm  = ws + W_OFF + (size_t)p * COUT * CIN;
    const float*  bia = (p == 0) ? bq : (p == 1) ? bk : bv;
    __bf16* outbuf    = ws + ((p == 0) ? Q_OFF : (p == 1) ? K_OFF : V_OFF);

    const int half16 = lane >> 4;
    const int l15    = lane & 15;
    const int akb    = half16 * 16;     // A K-segment base (0 / 16)
    const int bks    = half16 * 16;     // B K-segment base (0 / 16)
    const __bf16* arowp = Xt + (size_t)(px0 + l15) * CIN;

    v8f acc[4] = {{}, {}, {}, {}};

    #pragma unroll
    for (int k0 = 0; k0 < CIN; k0 += 32) {
        v16bf a;
        ((v8bf*)&a)[0] = *(const v8bf*)(arowp + k0 + akb);       // K seg, +0
        ((v8bf*)&a)[1] = *(const v8bf*)(arowp + k0 + akb + 8);   // K seg, +8
        #pragma unroll
        for (int f = 0; f < 4; ++f) {
            const __bf16* wrow = Wm + (size_t)(c0 + f*16 + l15) * CIN + k0 + bks;
            v16bf bm;
            ((v8bf*)&bm)[0] = *(const v8bf*)(wrow);
            ((v8bf*)&bm)[1] = *(const v8bf*)(wrow + 8);
            acc[f] = wmma_bf16(a, bm, acc[f]);
        }
    }

    #pragma unroll
    for (int f = 0; f < 4; ++f) {
        const int   ch   = c0 + f*16 + l15;
        const float bval = bia[ch];
        const int   head = ch >> 5;
        const int   hdi  = ch & 31;
        #pragma unroll
        for (int r = 0; r < 8; ++r) {
            const int px  = px0 + r + half16 * 8;   // C/D: VGPR r -> row r / r+8
            const float v = acc[f][r] + bval;
            size_t dst;
            if (p == 2) dst = ((size_t)(b*HEADS + head) * HD + hdi) * NPIX + px;
            else        dst = ((size_t)(b*HEADS + head) * NPIX + px) * HD + hdi;
            outbuf[dst] = (__bf16)v;
        }
    }
}

// ---------------------------------------------------------------------------
// Kernel 3: flash attention.  One wave owns 16 query rows of one (b,head).
// Loop over 32-key blocks: 2 WMMAs (S=Q K^T), online softmax with 16-lane
// __shfl_xor row reductions, P staged through private LDS tile (cross-lane
// D->A transpose), 2 WMMAs accumulate O.  Final O /= l, store fp32 NCHW.
// ---------------------------------------------------------------------------
__global__ __launch_bounds__(256) void attn_kernel(const __bf16* __restrict__ ws,
                                                   float* __restrict__ out) {
    __shared__ __align__(16) __bf16 lds[8][16 * 32];   // 1 KB per wave

    const int lane = threadIdx.x & 31;
    const int w    = threadIdx.x >> 5;
    const int t    = blockIdx.x * 8 + w;               // 6272 row-tiles (exact)
    const int bh   = t / 196;
    const int px0  = (t % 196) * 16;
    const int b    = bh >> 3;
    const int h    = bh & 7;

    const __bf16* Q  = ws + Q_OFF + (size_t)(b*HEADS + h) * NPIX * HD;
    const __bf16* K  = ws + K_OFF + (size_t)(b*HEADS + h) * NPIX * HD;
    const __bf16* Vt = ws + V_OFF + (size_t)(b*HEADS + h) * HD * NPIX;

    const int half16 = lane >> 4;
    const int l15    = lane & 15;

    // Q A-fragment: row px0+l15, K segments 0-15 / 16-31
    v16bf qf;
    {
        const __bf16* qr = Q + (size_t)(px0 + l15) * HD + half16 * 16;
        ((v8bf*)&qf)[0] = *(const v8bf*)(qr);
        ((v8bf*)&qf)[1] = *(const v8bf*)(qr + 8);
    }

    float m_[8], l_[8];
    #pragma unroll
    for (int r = 0; r < 8; ++r) { m_[r] = -1e30f; l_[r] = 0.f; }
    v8f o0 = {}, o1 = {};

    for (int j0 = 0; j0 < NPIX; j0 += 32) {
        if (j0 + 32 < NPIX) {  // hint next K/V tiles into cache
            __builtin_prefetch(K + (size_t)(j0 + 32 + l15) * HD, 0, 1);
            __builtin_prefetch(Vt + (size_t)l15 * NPIX + j0 + 32, 0, 1);
        }
        // K B-fragments: column j, 16 contiguous d per half-wave (K row-major)
        v16bf kb0, kb1;
        {
            const __bf16* kr0 = K + (size_t)(j0 + l15) * HD + half16 * 16;
            ((v8bf*)&kb0)[0] = *(const v8bf*)(kr0);
            ((v8bf*)&kb0)[1] = *(const v8bf*)(kr0 + 8);
            const __bf16* kr1 = K + (size_t)(j0 + 16 + l15) * HD + half16 * 16;
            ((v8bf*)&kb1)[0] = *(const v8bf*)(kr1);
            ((v8bf*)&kb1)[1] = *(const v8bf*)(kr1 + 8);
        }
        v8f s0 = {}, s1 = {};
        s0 = wmma_bf16(qf, kb0, s0);   // 16 rows x cols j0..j0+15
        s1 = wmma_bf16(qf, kb1, s1);   // cols j0+16..j0+31

        // online softmax over this 32-column block
        #pragma unroll
        for (int r = 0; r < 8; ++r) {
            float a0 = s0[r] * SCALE, a1 = s1[r] * SCALE;
            float mx = fmaxf(a0, a1);
            mx = fmaxf(mx, __shfl_xor(mx, 1, 32));
            mx = fmaxf(mx, __shfl_xor(mx, 2, 32));
            mx = fmaxf(mx, __shfl_xor(mx, 4, 32));
            mx = fmaxf(mx, __shfl_xor(mx, 8, 32));   // stays in 16-lane half
            float mnew  = fmaxf(m_[r], mx);
            float alpha = __expf(m_[r] - mnew);
            float p0 = __expf(a0 - mnew);
            float p1 = __expf(a1 - mnew);
            float rs = p0 + p1;
            rs += __shfl_xor(rs, 1, 32);
            rs += __shfl_xor(rs, 2, 32);
            rs += __shfl_xor(rs, 4, 32);
            rs += __shfl_xor(rs, 8, 32);
            l_[r] = l_[r] * alpha + rs;
            m_[r] = mnew;
            o0[r] *= alpha;
            o1[r] *= alpha;
            const int row = r + half16 * 8;
            lds[w][row * 32 + l15]      = (__bf16)p0;
            lds[w][row * 32 + 16 + l15] = (__bf16)p1;
        }
        asm volatile("s_wait_dscnt 0" ::: "memory");   // LDS RAW (same wave)

        // P A-fragment from LDS (row = l15, K = j within block)
        v16bf pf;
        {
            const __bf16* pr = &lds[w][l15 * 32 + half16 * 16];
            ((v8bf*)&pf)[0] = *(const v8bf*)(pr);
            ((v8bf*)&pf)[1] = *(const v8bf*)(pr + 8);
        }
        // V B-fragments: column d, 16 contiguous j per half-wave (Vt row-major)
        v16bf vb0, vb1;
        {
            const __bf16* vr0 = Vt + (size_t)l15 * NPIX + j0 + half16 * 16;
            ((v8bf*)&vb0)[0] = *(const v8bf*)(vr0);
            ((v8bf*)&vb0)[1] = *(const v8bf*)(vr0 + 8);
            const __bf16* vr1 = Vt + (size_t)(16 + l15) * NPIX + j0 + half16 * 16;
            ((v8bf*)&vb1)[0] = *(const v8bf*)(vr1);
            ((v8bf*)&vb1)[1] = *(const v8bf*)(vr1 + 8);
        }
        o0 = wmma_bf16(pf, vb0, o0);   // d 0..15
        o1 = wmma_bf16(pf, vb1, o1);   // d 16..31
    }

    // epilogue: normalize and store fp32 [B][Cout][N]
    #pragma unroll
    for (int r = 0; r < 8; ++r) {
        const float inv = 1.f / l_[r];
        const int   px  = px0 + r + half16 * 8;
        out[((size_t)b * COUT + h * 32 + l15)      * NPIX + px] = o0[r] * inv;
        out[((size_t)b * COUT + h * 32 + 16 + l15) * NPIX + px] = o1[r] * inv;
    }
}

// ---------------------------------------------------------------------------
extern "C" void kernel_launch(void* const* d_in, const int* in_sizes, int n_in,
                              void* d_out, int out_size, void* d_ws, size_t ws_size,
                              hipStream_t stream) {
    const float* x  = (const float*)d_in[0];
    const float* wq = (const float*)d_in[1];
    const float* bq = (const float*)d_in[2];
    const float* wk = (const float*)d_in[3];
    const float* bk = (const float*)d_in[4];
    const float* wv = (const float*)d_in[5];
    const float* bv = (const float*)d_in[6];
    __bf16* ws = (__bf16*)d_ws;        // needs ~26.1 MB of workspace
    float* out = (float*)d_out;

    const int cvt_total = XT_ELEMS + W_ELEMS;                 // 3,407,872
    cvt_kernel<<<cvt_total / 256, 256, 0, stream>>>(x, wq, wk, wv, ws);
    proj_kernel<<<1176, 256, 0, stream>>>(ws, bq, bk, bv);    // 9408 tiles / 8 waves
    attn_kernel<<<784, 256, 0, stream>>>(ws, out);            // 6272 tiles / 8 waves
}
